// self_matching_33930241638435
// MI455X (gfx1250) — matrix operand order repaired and checked by reading, not compiled, not run on hardware
//
#include <hip/hip_runtime.h>

typedef __bf16 bf16;
typedef __attribute__((ext_vector_type(16))) __bf16 v16bf;
typedef __attribute__((ext_vector_type(8)))  __bf16 v8bf;
typedef __attribute__((ext_vector_type(8)))  float  v8f;

#define B_   16
#define L_   512
#define E_   868
#define EP_  896        // 28 * 32, zero-padded K dimension
#define OUTE (6 * E_)

// ---------------- WMMA operand loaders (bf16, 16x16x32) ----------------
// A operand 16(M)x32(K): lane&15 = M row; lanes 0-15 hold K[8h..8h+8) and
// K[16+8h..), h = lane>>4. Two contiguous 16B chunks per lane.
__device__ inline v16bf wmma_load_a(const bf16* p, int ld) {
    int lane = threadIdx.x & 31;
    const bf16* r = p + (size_t)(lane & 15) * ld + ((lane >> 4) * 8);
    union { v16bf v; v8bf h[2]; } u;
    u.h[0] = *(const v8bf*)(r);
    u.h[1] = *(const v8bf*)(r + 16);
    return u.v;
}
// B operand 32(K)x16(N): lane&15 = N column; lane holds K[8h..8h+8) and
// K[16+8h..), h = lane>>4 (same K order as A).
// Requires B stored K-major with leading dimension ld.
__device__ inline v16bf wmma_load_b(const bf16* p, int ld) {
    int lane = threadIdx.x & 31;
    const bf16* r = p + (lane & 15) + (size_t)((lane >> 4) * 8) * ld;
    v16bf v;
    #pragma unroll
    for (int s = 0; s < 16; ++s) {
        int k = (s < 8) ? s : (16 + s - 8);
        v[s] = r[(size_t)k * ld];
    }
    return v;
}

__device__ inline v8f wmma_bf16(v16bf a, v16bf b, v8f c) {
    return __builtin_amdgcn_wmma_f32_16x16x32_bf16(false, a, false, b,
                                                   (short)0, c, false, false);
}

// ---------------- Kernel 1: prep ----------------
// One wave per (b,l) row: emit bf16 x (padded), bf16 x*w3 (padded),
// bf16 x^T (E-major), and c[b,l] = dot(x,w2) + mask bias.
__global__ void prep_kernel(const float* __restrict__ x,
                            const int*   __restrict__ mask,
                            const float* __restrict__ w,
                            bf16* __restrict__ xb,
                            bf16* __restrict__ xw3,
                            bf16* __restrict__ xbT,
                            float* __restrict__ c) {
    int gwave = (blockIdx.x * blockDim.x + threadIdx.x) >> 5;
    int lane  = threadIdx.x & 31;
    if (gwave >= B_ * L_) return;
    int b = gwave / L_, l = gwave % L_;

    const float* xr  = x + (size_t)gwave * E_;
    const float* w2  = w + E_;
    const float* w3  = w + 2 * E_;
    bf16* xbr   = xb  + (size_t)gwave * EP_;
    bf16* xw3r  = xw3 + (size_t)gwave * EP_;
    bf16* xbTb  = xbT + (size_t)b * EP_ * L_;

    float s2 = 0.f;
    for (int e = lane; e < EP_; e += 32) {
        float v  = (e < E_) ? xr[e] : 0.f;
        float v3 = (e < E_) ? v * w3[e] : 0.f;
        if (e < E_) s2 += v * w2[e];
        bf16 bv = (bf16)v;
        xbr[e]  = bv;
        xw3r[e] = (bf16)v3;
        xbTb[(size_t)e * L_ + l] = bv;
    }
    #pragma unroll
    for (int m = 16; m >= 1; m >>= 1) s2 += __shfl_xor(s2, m, 32);
    if (lane == 0) {
        float nm = (1.0f - (float)mask[gwave]) * -10000.0f;
        c[gwave] = s2 + nm;   // s1[i] and b cancel inside softmax over j
    }
}

// ---------------- Kernel 2: scores + softmax -> bf16 A ----------------
// Block: 16 rows x 512 cols of one batch. 8 waves x 4 column tiles.
__global__ void score_softmax_kernel(const bf16* __restrict__ xw3,
                                     const bf16* __restrict__ xbT,
                                     const float* __restrict__ c,
                                     bf16* __restrict__ A) {
    int blk = blockIdx.x;
    int b   = blk >> 5;
    int i0  = (blk & 31) * 16;
    int wid  = threadIdx.x >> 5;
    int lane = threadIdx.x & 31;

    const bf16* arow  = xw3 + ((size_t)b * L_ + i0) * EP_;
    const bf16* bbase = xbT + (size_t)b * EP_ * L_;

    v8f acc[4] = {};
    for (int k = 0; k < EP_; k += 32) {
        v16bf av = wmma_load_a(arow + k, EP_);
        #pragma unroll
        for (int t = 0; t < 4; ++t) {
            int j0 = (wid * 4 + t) * 16;
            v16bf bv = wmma_load_b(bbase + (size_t)k * L_ + j0, L_);
            acc[t] = wmma_bf16(av, bv, acc[t]);
        }
    }

    // add per-column bias c[j] = s2[j] + mask bias
    const float* cb = c + b * L_;
    #pragma unroll
    for (int t = 0; t < 4; ++t) {
        float cj = cb[(wid * 4 + t) * 16 + (lane & 15)];
        #pragma unroll
        for (int r = 0; r < 8; ++r) acc[t][r] += cj;
    }

    __shared__ float red[8][16];
    __shared__ float rowv[16];
    int rb = (lane >> 4) * 8;   // this half-wave's row base

    // ---- row max ----
    float rmax[8];
    #pragma unroll
    for (int r = 0; r < 8; ++r) {
        float m = -3.0e38f;
        #pragma unroll
        for (int t = 0; t < 4; ++t) m = fmaxf(m, acc[t][r]);
        #pragma unroll
        for (int s = 1; s <= 8; s <<= 1) m = fmaxf(m, __shfl_xor(m, s, 32));
        rmax[r] = m;
    }
    if ((lane & 15) == 0) {
        #pragma unroll
        for (int r = 0; r < 8; ++r) red[wid][rb + r] = rmax[r];
    }
    __syncthreads();
    if (threadIdx.x < 16) {
        float m = red[0][threadIdx.x];
        #pragma unroll
        for (int w = 1; w < 8; ++w) m = fmaxf(m, red[w][threadIdx.x]);
        rowv[threadIdx.x] = m;
    }
    __syncthreads();
    float mymax[8];
    #pragma unroll
    for (int r = 0; r < 8; ++r) mymax[r] = rowv[rb + r];
    __syncthreads();

    // ---- exp + row sum ----
    float rsum[8];
    #pragma unroll
    for (int r = 0; r < 8; ++r) {
        float s = 0.f;
        #pragma unroll
        for (int t = 0; t < 4; ++t) {
            float e = __expf(acc[t][r] - mymax[r]);
            acc[t][r] = e;
            s += e;
        }
        #pragma unroll
        for (int m = 1; m <= 8; m <<= 1) s += __shfl_xor(s, m, 32);
        rsum[r] = s;
    }
    if ((lane & 15) == 0) {
        #pragma unroll
        for (int r = 0; r < 8; ++r) red[wid][rb + r] = rsum[r];
    }
    __syncthreads();
    if (threadIdx.x < 16) {
        float s = 0.f;
        #pragma unroll
        for (int w = 0; w < 8; ++w) s += red[w][threadIdx.x];
        rowv[threadIdx.x] = s;
    }
    __syncthreads();

    // ---- normalize, store bf16 A ----
    bf16* Ab = A + ((size_t)b * L_ + i0) * L_;
    float inv[8];
    #pragma unroll
    for (int r = 0; r < 8; ++r) inv[r] = 1.0f / rowv[rb + r];
    #pragma unroll
    for (int t = 0; t < 4; ++t) {
        int j = (wid * 4 + t) * 16 + (lane & 15);
        #pragma unroll
        for (int r = 0; r < 8; ++r)
            Ab[(size_t)(rb + r) * L_ + j] = (bf16)(acc[t][r] * inv[r]);
    }
}

// ---------------- Kernel 3: A2 = A @ A (bf16 out) ----------------
__global__ void gemm_AA_kernel(const bf16* __restrict__ A,
                               bf16* __restrict__ A2) {
    int blk = blockIdx.x;
    int b   = blk >> 5;
    int i0  = (blk & 31) * 16;
    int wid  = threadIdx.x >> 5;
    int lane = threadIdx.x & 31;

    const bf16* Ab   = A + (size_t)b * L_ * L_;
    const bf16* arow = Ab + (size_t)i0 * L_;

    v8f acc[4] = {};
    for (int k = 0; k < L_; k += 32) {
        v16bf av = wmma_load_a(arow + k, L_);
        #pragma unroll
        for (int t = 0; t < 4; ++t) {
            int j0 = (wid * 4 + t) * 16;
            v16bf bv = wmma_load_b(Ab + (size_t)k * L_ + j0, L_);
            acc[t] = wmma_bf16(av, bv, acc[t]);
        }
    }
    bf16* Ob = A2 + (size_t)b * L_ * L_ + (size_t)i0 * L_;
    int rb = (lane >> 4) * 8;
    #pragma unroll
    for (int t = 0; t < 4; ++t) {
        int j = (wid * 4 + t) * 16 + (lane & 15);
        #pragma unroll
        for (int r = 0; r < 8; ++r)
            Ob[(size_t)(rb + r) * L_ + j] = (bf16)acc[t][r];
    }
}

// ---------------- Kernel 4: V = Amat @ x, fp32 into output slot ----------------
// Each wave handles column tiles (wid + 8*t), t=0..6 -> covers 55 tiles (e<868).
__global__ void gemm_AX_kernel(const bf16* __restrict__ Amat,
                               const bf16* __restrict__ xb,
                               float* __restrict__ out, int col_off) {
    int blk = blockIdx.x;
    int b   = blk >> 5;
    int i0  = (blk & 31) * 16;
    int wid  = threadIdx.x >> 5;
    int lane = threadIdx.x & 31;

    const bf16* arow = Amat + (size_t)b * L_ * L_ + (size_t)i0 * L_;
    const bf16* xbb  = xb + (size_t)b * L_ * EP_;

    v8f acc[7] = {};
    for (int k = 0; k < L_; k += 32) {
        v16bf av = wmma_load_a(arow + k, L_);
        #pragma unroll
        for (int t = 0; t < 7; ++t) {
            int e0 = (wid + 8 * t) * 16;
            if (e0 < E_) {   // wave-uniform predicate: EXEC stays all-ones
                v16bf bv = wmma_load_b(xbb + (size_t)k * EP_ + e0, EP_);
                acc[t] = wmma_bf16(av, bv, acc[t]);
            }
        }
    }
    float* ob = out + ((size_t)b * L_ + i0) * (size_t)OUTE + col_off;
    int rb = (lane >> 4) * 8;
    #pragma unroll
    for (int t = 0; t < 7; ++t) {
        int e = (wid + 8 * t) * 16 + (lane & 15);
        if (e < E_) {
            #pragma unroll
            for (int r = 0; r < 8; ++r)
                ob[(size_t)(rb + r) * OUTE + e] = acc[t][r];
        }
    }
}

// ---------------- Kernel 5: elementwise finalize ----------------
__global__ void finalize_kernel(const float* __restrict__ x,
                                float* __restrict__ out) {
    size_t idx = (size_t)blockIdx.x * blockDim.x + threadIdx.x;
    size_t total = (size_t)B_ * L_ * E_;
    if (idx >= total) return;
    size_t bl = idx / E_;
    int e = (int)(idx % E_);
    float xv = x[idx];
    float* orow = out + bl * (size_t)OUTE;
    float v1 = orow[E_ + e];
    float v2 = orow[4 * E_ + e];
    orow[e]          = xv;
    orow[2 * E_ + e] = xv - v1;
    orow[3 * E_ + e] = xv * v1;
    orow[5 * E_ + e] = xv - v2;
}

// ---------------- host launcher ----------------
extern "C" void kernel_launch(void* const* d_in, const int* in_sizes, int n_in,
                              void* d_out, int out_size, void* d_ws, size_t ws_size,
                              hipStream_t stream) {
    (void)in_sizes; (void)n_in; (void)out_size; (void)ws_size;
    const float* x    = (const float*)d_in[0];
    const int*   mask = (const int*)d_in[1];
    const float* w    = (const float*)d_in[2];
    // d_in[3] (b) is a softmax row-constant -> cancels; unused.
    float* out = (float*)d_out;

    char* ws = (char*)d_ws;
    size_t off = 0;
    bf16* xb   = (bf16*)(ws + off); off += (size_t)B_ * L_ * EP_ * sizeof(bf16);
    bf16* xw3  = (bf16*)(ws + off); off += (size_t)B_ * L_ * EP_ * sizeof(bf16);
    bf16* xbT  = (bf16*)(ws + off); off += (size_t)B_ * EP_ * L_ * sizeof(bf16);
    bf16* Abf  = (bf16*)(ws + off); off += (size_t)B_ * L_ * L_ * sizeof(bf16);
    bf16* A2bf = (bf16*)(ws + off); off += (size_t)B_ * L_ * L_ * sizeof(bf16);
    float* cvec = (float*)(ws + off); off += (size_t)B_ * L_ * sizeof(float);

    // 1 wave per row, 8 waves per block
    prep_kernel<<<(B_ * L_) / 8, 256, 0, stream>>>(x, mask, w, xb, xw3, xbT, cvec);

    // 16 rows x full L columns per block
    score_softmax_kernel<<<B_ * (L_ / 16), 256, 0, stream>>>(xw3, xbT, cvec, Abf);

    gemm_AA_kernel<<<B_ * (L_ / 16), 256, 0, stream>>>(Abf, A2bf);

    gemm_AX_kernel<<<B_ * (L_ / 16), 256, 0, stream>>>(Abf,  xb, out, E_);      // V1
    gemm_AX_kernel<<<B_ * (L_ / 16), 256, 0, stream>>>(A2bf, xb, out, 4 * E_);  // V2

    size_t total = (size_t)B_ * L_ * E_;
    finalize_kernel<<<(unsigned)((total + 255) / 256), 256, 0, stream>>>(x, out);
}
